// AttentionType1_4930622456071
// MI455X (gfx1250) — compile-verified, matrix operand order checked by reading, not run on hardware
//
#include <hip/hip_runtime.h>

// ---------------------------------------------------------------------------
// MI455X (gfx1250) fused multi-head attention, bf16 WMMA + fp32 accumulate.
// S=1024, E=1024, H=16, HD=64.
//
// Workspace layout:
//   Qb   bf16 [16][1024][64]   @ 0        (2 MB)  new_q = q@Wq^T + k_enc[0]
//   Kb   bf16 [16][1024][64]   @ 2 MB     (2 MB)
//   Vt   bf16 [16][64][1024]   @ 4 MB     (2 MB)  V transposed per head
//   dots f32  [16][1024][2]    @ 6 MB     (128 KB)
//   attnb bf16 [1024][1024]    @ 6.125 MB (2 MB)
// ---------------------------------------------------------------------------

// Flip to 0 if the assembler rejects the async-to-LDS mnemonic.
#define USE_ASYNC_LDS 1

typedef __attribute__((ext_vector_type(16))) __bf16        v16bf;
typedef __attribute__((ext_vector_type(8)))  float         v8f;
typedef __attribute__((ext_vector_type(4)))  unsigned int  u32x4;
typedef __attribute__((ext_vector_type(4)))  float         f32x4;

union Frag { u32x4 u[2]; v16bf h; };   // 16 bf16 = 8 VGPRs, loaded as 2x b128

__device__ __forceinline__ v8f wmma_bf16(v16bf a, v16bf b, v8f c) {
  // (neg_a, A, neg_b, B, c_mod, C, reuse_a, reuse_b)
  return __builtin_amdgcn_wmma_f32_16x16x32_bf16(false, a, false, b,
                                                 (short)0, c, false, false);
}

__device__ __forceinline__ unsigned short f2bf(float f) {
  unsigned int u = __float_as_uint(f);
  u += 0x7FFFu + ((u >> 16) & 1u);      // round-to-nearest-even
  return (unsigned short)(u >> 16);
}
__device__ __forceinline__ float bf2f(unsigned short h) {
  return __uint_as_float(((unsigned int)h) << 16);
}

// 16-byte global -> LDS copy. Async variant uses the CDNA5 Tensor/async path
// (GLOBAL_LOAD_ASYNC_TO_LDS_B128, tracked by ASYNCcnt; LDS offset is the low
// 32 bits of the flat shared-aperture address per ISA 10.2 aperture rules).
__device__ __forceinline__ void cp_b128(void* lds_dst, const void* gsrc) {
#if USE_ASYNC_LDS
  asm volatile("global_load_async_to_lds_b128 %0, %1, off"
               :: "v"((unsigned int)(unsigned long long)lds_dst),
                  "v"((unsigned long long)gsrc)
               : "memory");
#else
  *(u32x4*)lds_dst = *(const u32x4*)gsrc;
#endif
}
__device__ __forceinline__ void cp_wait() {
#if USE_ASYNC_LDS
  asm volatile("s_wait_asynccnt 0x0" ::: "memory");
#endif
}

// ---------------------------------------------------------------------------
// 1024x1024x1024 GEMM  C = A @ W^T  (both operands contiguous along K).
// Block = 256 threads = 8 waves arranged 2(M) x 4(N); block tile 128x256;
// wave tile 64x64 -> 16 WMMA per K-step of 32 vs 16 fragment b128 loads.
// MODE 0: A=f32 q, epilogue += k_enc[0][n], write bf16 Qb[h][s][d]
// MODE 1: A=f32 k, write bf16 Kb[h][s][d]
// MODE 2: A=f32 v, write bf16 Vt[h][d][s] (transposed, packed b128 stores)
// MODE 3: A=bf16 attn, W=Wo, write f32 out[s][e]
// ---------------------------------------------------------------------------
template<int MODE>
__global__ __launch_bounds__(256)
void gemm_kernel(const float* __restrict__ A32,
                 const unsigned short* __restrict__ A16,
                 const float* __restrict__ W,
                 const float* __restrict__ kenc,
                 unsigned short* __restrict__ outb,
                 float* __restrict__ outf)
{
  __shared__ unsigned short As[128 * 32];   //  8 KB
  __shared__ unsigned short Ws[256 * 32];   // 16 KB

  const int tid  = threadIdx.x;
  const int lane = tid & 31;
  const int wave = tid >> 5;
  const int half = lane >> 4;      // K-chunk select (lane half)
  const int lcol = lane & 15;      // row (A) / column (B) within fragment
  const int waveM = (wave & 1) * 64;        // 0 or 64
  const int waveN = (wave >> 1) * 64;       // 0,64,128,192
  const int m0 = blockIdx.y * 128;
  const int n0 = blockIdx.x * 256;

  v8f acc[4][4] = {};

  const int sr = tid >> 1;          // A staging row 0..127
  const int sc = (tid & 1) * 16;    // A staging col 0 or 16

  for (int k0 = 0; k0 < 1024; k0 += 32) {
    // ---- stage A tile (128x32) ----
    if (MODE == 3) {
      const unsigned short* src = A16 + (m0 + sr) * 1024 + k0 + sc;
      cp_b128(As + sr * 32 + sc,     src);
      cp_b128(As + sr * 32 + sc + 8, src + 8);
      if (k0 + 32 < 1024) __builtin_prefetch(src + 32, 0, 0);
    } else {
      const f32x4* src = (const f32x4*)(A32 + (m0 + sr) * 1024 + k0 + sc);
      if (k0 + 32 < 1024) __builtin_prefetch((const float*)src + 32, 0, 0);
      alignas(16) unsigned short tmp[16];
#pragma unroll
      for (int i = 0; i < 4; ++i) {
        f32x4 f = src[i];
        tmp[4*i+0] = f2bf(f[0]); tmp[4*i+1] = f2bf(f[1]);
        tmp[4*i+2] = f2bf(f[2]); tmp[4*i+3] = f2bf(f[3]);
      }
      u32x4* dst = (u32x4*)(As + sr * 32 + sc);
      dst[0] = ((const u32x4*)tmp)[0];
      dst[1] = ((const u32x4*)tmp)[1];
    }
    // ---- stage W tile (256x32, rows = output columns) ----
    {
      const f32x4* src = (const f32x4*)(W + (n0 + tid) * 1024 + k0);
      if (k0 + 32 < 1024) __builtin_prefetch((const float*)src + 32, 0, 0);
      alignas(16) unsigned short tmp[32];
#pragma unroll
      for (int i = 0; i < 8; ++i) {
        f32x4 f = src[i];
        tmp[4*i+0] = f2bf(f[0]); tmp[4*i+1] = f2bf(f[1]);
        tmp[4*i+2] = f2bf(f[2]); tmp[4*i+3] = f2bf(f[3]);
      }
      u32x4* dst = (u32x4*)(Ws + tid * 32);
#pragma unroll
      for (int i = 0; i < 4; ++i) dst[i] = ((const u32x4*)tmp)[i];
    }
    if (MODE == 3) cp_wait();
    __syncthreads();

    // ---- fragment loads + 16 WMMA ----
    Frag af[4], bfr[4];
#pragma unroll
    for (int fm = 0; fm < 4; ++fm) {
      const unsigned short* p = As + (waveM + fm * 16 + lcol) * 32 + half * 8;
      af[fm].u[0] = ((const u32x4*)p)[0];        // K 0..7  / 8..15
      af[fm].u[1] = ((const u32x4*)(p + 16))[0]; // K 16..23 / 24..31
    }
#pragma unroll
    for (int fn = 0; fn < 4; ++fn) {
      const unsigned short* p = Ws + (waveN + fn * 16 + lcol) * 32 + half * 8;
      bfr[fn].u[0] = ((const u32x4*)p)[0];
      bfr[fn].u[1] = ((const u32x4*)(p + 16))[0];
    }
#pragma unroll
    for (int fm = 0; fm < 4; ++fm)
#pragma unroll
      for (int fn = 0; fn < 4; ++fn)
        acc[fm][fn] = wmma_bf16(af[fm].h, bfr[fn].h, acc[fm][fn]);
    __syncthreads();
  }

  // ---- epilogue ----
#pragma unroll
  for (int fm = 0; fm < 4; ++fm) {
#pragma unroll
    for (int fn = 0; fn < 4; ++fn) {
      if (MODE == 2) {
        // Vt[h][d][s]: register index v walks consecutive s -> packed store.
        const int n  = n0 + waveN + fn * 16 + lcol;
        const int hh = n >> 6, d = n & 63;
        const int mbase = m0 + waveM + fm * 16 + half * 8;
        alignas(16) unsigned short tmp[8];
#pragma unroll
        for (int v = 0; v < 8; ++v) tmp[v] = f2bf(acc[fm][fn][v]);
        *(u32x4*)(outb + (hh << 16) + (d << 10) + mbase) = ((const u32x4*)tmp)[0];
      } else {
#pragma unroll
        for (int v = 0; v < 8; ++v) {
          const int m = m0 + waveM + fm * 16 + half * 8 + v;
          const int n = n0 + waveN + fn * 16 + lcol;
          float val = acc[fm][fn][v];
          if (MODE == 0) val += kenc[n];          // q_emb = k_enc[0] broadcast
          if (MODE == 3) {
            outf[m * 1024 + n] = val;
          } else {
            const int hh = n >> 6, d = n & 63;
            outb[(hh << 16) + (m << 6) + d] = f2bf(val);
          }
        }
      }
    }
  }
}

// ---------------------------------------------------------------------------
// dots[h][s][v] = sum_d new_q[h][s][d] * k_enc[v][h*64+d]
// ---------------------------------------------------------------------------
__global__ __launch_bounds__(256)
void dots_kernel(const unsigned short* __restrict__ Qb,
                 const float* __restrict__ kenc,
                 float* __restrict__ dots)
{
  const int idx = blockIdx.x * 256 + threadIdx.x;   // h*1024 + s
  if (idx >= 16 * 1024) return;
  const int h = idx >> 10, s = idx & 1023;
  const unsigned short* q = Qb + (h << 16) + (s << 6);
  const float* e0 = kenc + (h << 6);
  const float* e1 = kenc + 1024 + (h << 6);
  float a0 = 0.f, a1 = 0.f;
#pragma unroll 8
  for (int d = 0; d < 64; ++d) {
    const float qv = bf2f(q[d]);
    a0 += qv * e0[d];
    a1 += qv * e1[d];
  }
  dots[idx * 2 + 0] = a0;
  dots[idx * 2 + 1] = a1;
}

// ---------------------------------------------------------------------------
// Flash attention per (head, 128-query tile). 8 waves, 16 query rows each.
// Key tiles of 128: S = Q K^T (WMMA) -> +s2, scale, mask(1e-30) -> online
// softmax -> P (bf16 via per-wave LDS relayout) -> O += P V (WMMA).
// K/V tiles staged with async global->LDS (ASYNCcnt) + prefetch of next tile.
// ---------------------------------------------------------------------------
__global__ __launch_bounds__(256)
void attn_kernel(const unsigned short* __restrict__ Qb,
                 const unsigned short* __restrict__ Kb,
                 const unsigned short* __restrict__ Vt,
                 const float* __restrict__ dots,
                 const unsigned char* __restrict__ mask,  // bool (H,S,S)
                 const float* __restrict__ utt,
                 const int* __restrict__ spk,
                 unsigned short* __restrict__ attnb)
{
  __shared__ unsigned short Ks[128 * 64];     // keys  [t][d]   16 KB
  __shared__ unsigned short Vs[64 * 128];     // V^T   [d][t]   16 KB
  __shared__ unsigned short Ps[8 * 16 * 128]; // per-wave P     32 KB

  const int tid  = threadIdx.x;
  const int lane = tid & 31;
  const int wave = tid >> 5;
  const int half = lane >> 4;
  const int lcol = lane & 15;
  const int h  = blockIdx.y;
  const int q0 = blockIdx.x * 128;
  const int wq = wave * 16;
  unsigned short* Pw = Ps + wave * 16 * 128;

  // Q fragments (A operand), kept in registers for all key tiles.
  Frag qa[2];
  {
    const unsigned short* qp = Qb + (h << 16) + ((q0 + wq + lcol) << 6);
#pragma unroll
    for (int kf = 0; kf < 2; ++kf) {
      const unsigned short* p = qp + kf * 32 + half * 8;
      qa[kf].u[0] = ((const u32x4*)p)[0];
      qa[kf].u[1] = ((const u32x4*)(p + 16))[0];
    }
  }

  float d0[8], d1[8], mprev[8], lsum[8];
  v8f O[4] = {};
#pragma unroll
  for (int v = 0; v < 8; ++v) {
    const int r = q0 + wq + half * 8 + v;
    d0[v] = dots[((h << 10) + r) * 2 + 0];
    d1[v] = dots[((h << 10) + r) * 2 + 1];
    mprev[v] = -3.0e38f;
    lsum[v] = 0.f;
  }

  for (int t0 = 0; t0 < 1024; t0 += 128) {
    // ---- stage K tile and V^T tile (async global -> LDS) ----
    {
      const unsigned short* srcK = Kb + ((h << 10) + t0) * 64;
#pragma unroll
      for (int i = 0; i < 4; ++i) {
        const int idx = (tid + i * 256) * 8;          // elements (8 bf16/b128)
        cp_b128(Ks + idx, srcK + idx);
      }
#pragma unroll
      for (int i = 0; i < 4; ++i) {
        const int idx = tid + i * 256;
        const int d = idx >> 4, c = (idx & 15) << 3;
        cp_b128(Vs + d * 128 + c,
                Vt + (h << 6) * 1024 + d * 1024 + t0 + c);
      }
      if (t0 + 128 < 1024) {
        __builtin_prefetch(srcK + 128 * 64 + tid * 32, 0, 0);
        __builtin_prefetch(Vt + (h << 6) * 1024 + (tid >> 2) * 1024
                              + t0 + 128 + (tid & 3) * 32, 0, 0);
      }
      cp_wait();
    }
    __syncthreads();

    // ---- S = Q K^T : 8 column fragments, 2 WMMA each (K=64) ----
    v8f sf[8];
#pragma unroll
    for (int nf = 0; nf < 8; ++nf) {
      Frag b0, b1;
      const unsigned short* kp = Ks + (nf * 16 + lcol) * 64 + half * 8;
      b0.u[0] = ((const u32x4*)kp)[0];
      b0.u[1] = ((const u32x4*)(kp + 16))[0];
      b1.u[0] = ((const u32x4*)(kp + 32))[0];
      b1.u[1] = ((const u32x4*)(kp + 48))[0];
      v8f c = {};
      c = wmma_bf16(qa[0].h, b0.h, c);
      c = wmma_bf16(qa[1].h, b1.h, c);
      sf[nf] = c;
    }

    // ---- +s2, scale, mask (reference uses 1e-30, NOT -inf), row max ----
    float rmax[8];
#pragma unroll
    for (int v = 0; v < 8; ++v) rmax[v] = -3.0e38f;
#pragma unroll
    for (int nf = 0; nf < 8; ++nf) {
#pragma unroll
      for (int v = 0; v < 8; ++v) {
        const int row = q0 + wq + half * 8 + v;
        const int col = t0 + nf * 16 + lcol;
        const long pix = (long)row * 1024 + col;
        const float s2 = (spk[pix] ? d1[v] : d0[v]) * utt[pix];
        float sv = (sf[nf][v] + s2) * 0.125f;     // scaling = HD^-0.5
        if (mask[(long)h * 1048576 + pix]) sv = 1e-30f;
        sf[nf][v] = sv;
        rmax[v] = fmaxf(rmax[v], sv);
      }
    }
    // C-layout rows live across a 16-lane half: xor 1/2/4/8 reduces a row.
#pragma unroll
    for (int v = 0; v < 8; ++v)
#pragma unroll
      for (int m = 1; m < 16; m <<= 1)
        rmax[v] = fmaxf(rmax[v], __shfl_xor(rmax[v], m, 32));

    // ---- online softmax update ----
#pragma unroll
    for (int v = 0; v < 8; ++v) {
      const float mnew  = fmaxf(mprev[v], rmax[v]);
      const float scale = __expf(mprev[v] - mnew);
      mprev[v] = mnew;
      float ls = 0.f;
#pragma unroll
      for (int nf = 0; nf < 8; ++nf) {
        const float p = __expf(sf[nf][v] - mnew);
        sf[nf][v] = p;
        ls += p;
      }
#pragma unroll
      for (int m = 1; m < 16; m <<= 1) ls += __shfl_xor(ls, m, 32);
      lsum[v] = lsum[v] * scale + ls;
#pragma unroll
      for (int df = 0; df < 4; ++df) O[df][v] *= scale;
    }

    // ---- relayout P (C-format) -> A-format via per-wave LDS region ----
#pragma unroll
    for (int nf = 0; nf < 8; ++nf)
#pragma unroll
      for (int v = 0; v < 8; ++v)
        Pw[(half * 8 + v) * 128 + nf * 16 + lcol] = f2bf(sf[nf][v]);

    // ---- O += P V : 4 d-fragments x 4 K-chunks of 32 ----
#pragma unroll
    for (int kf = 0; kf < 4; ++kf) {
      Frag pa;
      const unsigned short* pp = Pw + lcol * 128 + kf * 32 + half * 8;
      pa.u[0] = ((const u32x4*)pp)[0];
      pa.u[1] = ((const u32x4*)(pp + 16))[0];
#pragma unroll
      for (int df = 0; df < 4; ++df) {
        Frag vb;
        const unsigned short* vp = Vs + (df * 16 + lcol) * 128 + kf * 32 + half * 8;
        vb.u[0] = ((const u32x4*)vp)[0];
        vb.u[1] = ((const u32x4*)(vp + 16))[0];
        O[df] = wmma_bf16(pa.h, vb.h, O[df]);
      }
    }
    __syncthreads();
  }

  // ---- normalize + store attn output (bf16, head-interleaved row-major) ----
#pragma unroll
  for (int df = 0; df < 4; ++df)
#pragma unroll
    for (int v = 0; v < 8; ++v) {
      const int row = q0 + wq + half * 8 + v;
      const float o = O[df][v] / lsum[v];
      attnb[row * 1024 + (h << 6) + df * 16 + lcol] = f2bf(o);
    }
}

// ---------------------------------------------------------------------------
extern "C" void kernel_launch(void* const* d_in, const int* in_sizes, int n_in,
                              void* d_out, int out_size, void* d_ws, size_t ws_size,
                              hipStream_t stream)
{
  const float*         q    = (const float*)d_in[0];
  const float*         k    = (const float*)d_in[1];
  const float*         v    = (const float*)d_in[2];
  const unsigned char* mask = (const unsigned char*)d_in[3]; // jax bool = 1 byte
  const float*         utt  = (const float*)d_in[4];
  const int*           spk  = (const int*)d_in[5];
  const float*         Wq   = (const float*)d_in[6];
  const float*         Wk   = (const float*)d_in[7];
  const float*         Wv   = (const float*)d_in[8];
  const float*         Wo   = (const float*)d_in[9];
  const float*         kenc = (const float*)d_in[10];
  float*               out  = (float*)d_out;

  unsigned short* Qb    = (unsigned short*)d_ws;
  unsigned short* Kb    = Qb + (1 << 20);
  unsigned short* Vt    = Kb + (1 << 20);
  float*          dotsb = (float*)(Vt + (1 << 20));
  unsigned short* attnb = (unsigned short*)(dotsb + 2 * 16 * 1024);

  const dim3 b(256);
  gemm_kernel<0><<<dim3(4, 8), b, 0, stream>>>(q, nullptr, Wq, kenc, Qb, nullptr);
  gemm_kernel<1><<<dim3(4, 8), b, 0, stream>>>(k, nullptr, Wk, kenc, Kb, nullptr);
  gemm_kernel<2><<<dim3(4, 8), b, 0, stream>>>(v, nullptr, Wv, kenc, Vt, nullptr);
  dots_kernel<<<dim3(64), b, 0, stream>>>(Qb, kenc, dotsb);
  attn_kernel<<<dim3(8, 16), b, 0, stream>>>(Qb, Kb, Vt, dotsb, mask, utt, spk, attnb);
  gemm_kernel<3><<<dim3(4, 8), b, 0, stream>>>(nullptr, attnb, Wo, kenc, nullptr, out);
}
